// GATv4_32710470926858
// MI455X (gfx1250) — compile-verified, matrix operand order checked by reading, not run on hardware
//
#include <hip/hip_runtime.h>
#include <hip/hip_bf16.h>
#include <math.h>

typedef __attribute__((ext_vector_type(16))) __bf16 v16bf;
typedef __attribute__((ext_vector_type(8)))  float  v8f;
typedef __attribute__((ext_vector_type(4)))  float  f4;

#define NEG_SLOPE 0.2f
#define NNODES 32768
#define INDIM 2048

__device__ __forceinline__ float leakyf(float v) { return v > 0.f ? v : NEG_SLOPE * v; }
__device__ __forceinline__ float eluf(float v)   { return v > 0.f ? v : (expf(v) - 1.f); }

__device__ __forceinline__ void atomicMaxF(float* addr, float val) {
    // classic signed-max / unsigned-min trick for IEEE-754 float max
    int iv = __float_as_int(val);
    if (iv >= 0) atomicMax((int*)addr, iv);
    else         atomicMin((unsigned int*)addr, (unsigned int)iv);
}

__device__ __forceinline__ v16bf packA(f4 a0, f4 a1, f4 a2, f4 a3) {
    v16bf A;
    A[0]  = (__bf16)a0.x; A[1]  = (__bf16)a0.y; A[2]  = (__bf16)a0.z; A[3]  = (__bf16)a0.w;
    A[4]  = (__bf16)a1.x; A[5]  = (__bf16)a1.y; A[6]  = (__bf16)a1.z; A[7]  = (__bf16)a1.w;
    A[8]  = (__bf16)a2.x; A[9]  = (__bf16)a2.y; A[10] = (__bf16)a2.z; A[11] = (__bf16)a2.w;
    A[12] = (__bf16)a3.x; A[13] = (__bf16)a3.y; A[14] = (__bf16)a3.z; A[15] = (__bf16)a3.w;
    return A;
}

// Pack a [K, ncols] f32 weight into fragment-ordered bf16:
// out[((kt*nTiles + nt)*32 + lane)*16 + e] = W[k(e,lane) * ncols + nt*16 + lane%16]
__global__ void pack_b_kernel(const float* __restrict__ W, __bf16* __restrict__ out,
                              int ncols, int nTiles, int kTiles) {
    int i = blockIdx.x * blockDim.x + threadIdx.x;
    if (i >= kTiles * nTiles * 512) return;
    int e    = i & 15;
    int lane = (i >> 4) & 31;
    int nt   = (i >> 9) % nTiles;
    int kt   = (i >> 9) / nTiles;
    int h    = lane >> 4;
    int col  = nt * 16 + (lane & 15);
    int k    = kt * 32 + (e < 8 ? 8 * h + e : 16 + 8 * h + (e - 8));
    out[i] = (__bf16)W[(size_t)k * ncols + col];
}

// ---------------- GEMM1: h1[32768,32] = x[32768,2048] @ W1[2048,32] ----------------
// 4 waves / block, each wave -> one 16x16 output tile. A: non-temporal streaming loads.
// B: prefragmented bf16 (one 32-byte contiguous load per lane per K-tile).
__global__ void __launch_bounds__(128)
gat_gemm1_wmma(const float* __restrict__ X, const __bf16* __restrict__ Wbf,
               float* __restrict__ Out) {
    const int lane  = threadIdx.x & 31;
    const int wave  = threadIdx.x >> 5;            // 0..3
    const int mBase = (blockIdx.x * 2 + (wave >> 1)) * 16;
    const int nTile = wave & 1;
    const int l16   = lane & 15;
    const int h     = lane >> 4;                   // K half selector
    const int col   = nTile * 16 + l16;

    v8f acc = {};
    const float* arow = X + (size_t)(mBase + l16) * INDIM + 8 * h;
    const __bf16* brow = Wbf + ((size_t)nTile * 32 + lane) * 16;
    for (int kt = 0; kt < INDIM / 32; ++kt) {
        const float* ap = arow + kt * 32;
        f4 a0 = __builtin_nontemporal_load((const f4*)(ap));
        f4 a1 = __builtin_nontemporal_load((const f4*)(ap + 4));
        f4 a2 = __builtin_nontemporal_load((const f4*)(ap + 16));
        f4 a3 = __builtin_nontemporal_load((const f4*)(ap + 20));
        v16bf A  = packA(a0, a1, a2, a3);
        v16bf Bf = *(const v16bf*)(brow + (size_t)kt * 2 * 512);

        acc = __builtin_amdgcn_wmma_f32_16x16x32_bf16(
            false, A, false, Bf, (short)0, acc, false, false);
    }
#pragma unroll
    for (int v = 0; v < 8; ++v)
        Out[(size_t)(mBase + v + 8 * h) * 32 + col] = acc[v];
}

// ---------------- GEMM2: h2[32768,48] = h1'[32768,32] @ W2[32,48] (K=32: 1 WMMA) ----
__global__ void __launch_bounds__(192)
gat_gemm2_wmma(const float* __restrict__ Hin, const __bf16* __restrict__ Wbf,
               float* __restrict__ Out) {
    const int lane  = threadIdx.x & 31;
    const int wave  = threadIdx.x >> 5;            // 0..5
    const int mBase = (blockIdx.x * 2 + wave / 3) * 16;
    const int nTile = wave % 3;
    const int l16   = lane & 15;
    const int h     = lane >> 4;
    const int col   = nTile * 16 + l16;

    const float* ap = Hin + (size_t)(mBase + l16) * 32 + 8 * h;
    f4 a0 = *(const f4*)(ap);
    f4 a1 = *(const f4*)(ap + 4);
    f4 a2 = *(const f4*)(ap + 16);
    f4 a3 = *(const f4*)(ap + 20);
    v16bf A  = packA(a0, a1, a2, a3);
    v16bf Bf = *(const v16bf*)(Wbf + ((size_t)nTile * 32 + lane) * 16);

    v8f acc = {};
    acc = __builtin_amdgcn_wmma_f32_16x16x32_bf16(
        false, A, false, Bf, (short)0, acc, false, false);
#pragma unroll
    for (int v = 0; v < 8; ++v)
        Out[(size_t)(mBase + v + 8 * h) * 48 + col] = acc[v];
}

// ---------------- attention scalar kernels ----------------
// Computes per-(node,head) src/dst coefficients AND seeds the per-dst running max
// with the self-loop edge value (self-loops always exist -> no empty segments).
__global__ void attn_coef_kernel(const float* __restrict__ feat,
                                 const float* __restrict__ aw_s,
                                 const float* __restrict__ aw_d,
                                 float* __restrict__ as_out, float* __restrict__ ad_out,
                                 float* __restrict__ m_out,
                                 int Hh, int C) {
    int i = blockIdx.x * blockDim.x + threadIdx.x;      // node*H + h
    if (i >= NNODES * Hh) return;
    int n = i / Hh, h = i % Hh;
    const float* f = feat + ((size_t)n * Hh + h) * C;
    float s1 = 0.f, s2 = 0.f;
    for (int c = 0; c < C; ++c) {
        float v = f[c];
        s1 += v * aw_s[h * C + c];
        s2 += v * aw_d[h * C + c];
    }
    as_out[i] = s1;
    ad_out[i] = s2;
    m_out[i]  = leakyf(s1 + s2);
}

__global__ void edge_max_kernel(const int* __restrict__ src, const int* __restrict__ dst,
                                const float* __restrict__ as_, const float* __restrict__ ad_,
                                float* __restrict__ m, int E, int Hh) {
    int e = blockIdx.x * blockDim.x + threadIdx.x;
    if (e >= E) return;
    int s = src[e], d = dst[e];
    for (int h = 0; h < Hh; ++h) {
        float v = leakyf(as_[s * Hh + h] + ad_[d * Hh + h]);
        atomicMaxF(&m[d * Hh + h], v);
    }
}

// seed s and acc with the self-loop contribution (fully initializes both buffers)
__global__ void init_s_acc_kernel(const float* __restrict__ feat,
                                  const float* __restrict__ as_, const float* __restrict__ ad_,
                                  const float* __restrict__ m,
                                  float* __restrict__ ssum, float* __restrict__ acc,
                                  int Hh, int C) {
    int i = blockIdx.x * blockDim.x + threadIdx.x;      // node*H + h
    if (i >= NNODES * Hh) return;
    int n = i / Hh, h = i % Hh;
    float p = expf(leakyf(as_[i] + ad_[i]) - m[i]);
    ssum[i] = p;
    const float* f = feat + ((size_t)n * Hh + h) * C;
    float* a = acc + ((size_t)n * Hh + h) * C;
    for (int c = 0; c < C; ++c) a[c] = p * f[c];
}

__global__ void edge_acc_kernel(const int* __restrict__ src, const int* __restrict__ dst,
                                const float* __restrict__ feat,
                                const float* __restrict__ as_, const float* __restrict__ ad_,
                                const float* __restrict__ m,
                                float* __restrict__ ssum, float* __restrict__ acc,
                                int E, int Hh, int C) {
    int e = blockIdx.x * blockDim.x + threadIdx.x;
    if (e >= E) return;
    int s = src[e], d = dst[e];
    for (int h = 0; h < Hh; ++h) {
        float p = expf(leakyf(as_[s * Hh + h] + ad_[d * Hh + h]) - m[d * Hh + h]);
        atomicAdd(&ssum[d * Hh + h], p);
        const float* f = feat + ((size_t)s * Hh + h) * C;
        float* a = acc + ((size_t)d * Hh + h) * C;
        for (int c = 0; c < C; ++c) atomicAdd(&a[c], p * f[c]);
    }
}

// out = elu(acc / (s + 1e-16) + bias), in place on acc
__global__ void finalize_kernel(float* __restrict__ acc, const float* __restrict__ ssum,
                                const float* __restrict__ bias, int Hh, int C) {
    int i = blockIdx.x * blockDim.x + threadIdx.x;
    int HC = Hh * C;
    if (i >= NNODES * HC) return;
    int n = i / HC, rem = i % HC, h = rem / C;
    float v = acc[i] / (ssum[n * Hh + h] + 1e-16f) + bias[rem];
    acc[i] = eluf(v);
}

// ---------------- pooling / layernorm / MLP ----------------
__global__ void pool2_kernel(const float* __restrict__ h2, const float* __restrict__ w,
                             const float* __restrict__ bscalar, float* __restrict__ xout) {
    int n = blockIdx.x * blockDim.x + threadIdx.x;
    if (n >= NNODES) return;
    const float* r = h2 + (size_t)n * 48;
    float acc = bscalar[0];
    for (int c = 0; c < 48; ++c) acc += r[c] * w[c];
    xout[n] = acc;
}

__global__ void __launch_bounds__(256)
layernorm_kernel(const float* __restrict__ x, const float* __restrict__ g,
                 const float* __restrict__ b, float* __restrict__ out) {
    __shared__ float ssum[256], ssq[256];
    const int r = blockIdx.x, t = threadIdx.x;
    const float* xr = x + (size_t)r * INDIM;
    float s = 0.f, q = 0.f;
    for (int i = t; i < INDIM; i += 256) { float v = xr[i]; s += v; q += v * v; }
    ssum[t] = s; ssq[t] = q;
    __syncthreads();
    for (int w = 128; w > 0; w >>= 1) {
        if (t < w) { ssum[t] += ssum[t + w]; ssq[t] += ssq[t + w]; }
        __syncthreads();
    }
    float mu   = ssum[0] / (float)INDIM;
    float var  = ssq[0] / (float)INDIM - mu * mu;
    float rstd = rsqrtf(var + 1e-5f);
    for (int i = t; i < INDIM; i += 256)
        out[(size_t)r * INDIM + i] = (xr[i] - mu) * rstd * g[i] + b[i];
}

__global__ void fc_kernel(const float* __restrict__ in, const float* __restrict__ W,
                          const float* __restrict__ bias, float* __restrict__ out,
                          int K, int M, int act) {
    int i = blockIdx.x * blockDim.x + threadIdx.x;      // row*M + col over 16*M
    if (i >= 16 * M) return;
    int r = i / M, j = i % M;
    const float* x = in + (size_t)r * K;
    float acc = bias[j];
    for (int k = 0; k < K; ++k) acc += x[k] * W[(size_t)k * M + j];
    out[i] = act ? eluf(acc) : acc;
}

__global__ void last_kernel(const float* __restrict__ enc, const float* __restrict__ w,
                            const float* __restrict__ bscalar, float* __restrict__ pred) {
    int r = threadIdx.x;
    if (r >= 16) return;
    const float* x = enc + (size_t)r * 128;
    float acc = bscalar[0];
    for (int k = 0; k < 128; ++k) acc += x[k] * w[k];
    pred[r] = 6.f / (1.f + expf(-acc)) - 3.f;
}

extern "C" void kernel_launch(void* const* d_in, const int* in_sizes, int n_in,
                              void* d_out, int out_size, void* d_ws, size_t ws_size,
                              hipStream_t stream) {
    const float* x       = (const float*)d_in[0];
    const int*   ei      = (const int*)d_in[1];
    const float* W1      = (const float*)d_in[2];
    const float* as1w    = (const float*)d_in[3];
    const float* ad1w    = (const float*)d_in[4];
    const float* b1      = (const float*)d_in[5];
    const float* W2      = (const float*)d_in[6];
    const float* as2w    = (const float*)d_in[7];
    const float* ad2w    = (const float*)d_in[8];
    const float* b2      = (const float*)d_in[9];
    const float* pool2_w = (const float*)d_in[12];
    const float* pool2_b = (const float*)d_in[13];
    const float* ln0_g   = (const float*)d_in[14];
    const float* ln0_b   = (const float*)d_in[15];
    const float* fc1_w   = (const float*)d_in[18];
    const float* fc1_b   = (const float*)d_in[19];
    const float* fc2_w   = (const float*)d_in[20];
    const float* fc2_b   = (const float*)d_in[21];
    const float* fc3_w   = (const float*)d_in[22];
    const float* fc3_b   = (const float*)d_in[23];
    const float* fc4_w   = (const float*)d_in[24];
    const float* fc4_b   = (const float*)d_in[25];
    const float* last_w  = (const float*)d_in[26];
    const float* last_b  = (const float*)d_in[27];

    float* out = (float*)d_out;           // [0,32768) multiscale | [32768,34816) encoded | [34816,34832) pred
    float* ws  = (float*)d_ws;
    const int E = in_sizes[1] / 2;

    // workspace layout (floats; all blocks 32B-aligned)
    float* h1   = ws;                     // 32768*32
    float* hp1  = h1  + 1048576;          // aggregated -> h1' (32768*32)
    float* as1  = hp1 + 1048576;          // 32768*4
    float* ad1  = as1 + 131072;
    float* m1   = ad1 + 131072;
    float* s1   = m1  + 131072;
    float* h2   = s1  + 131072;           // 32768*48
    float* hp2  = h2  + 1572864;          // aggregated -> h2' (32768*48)
    float* as2  = hp2 + 1572864;          // 32768*3
    float* ad2  = as2 + 98304;
    float* m2   = ad2 + 98304;
    float* s2   = m2  + 98304;
    float* x2b  = s2  + 98304;            // 16*2048
    float* e1   = x2b + 32768;            // 16*1024
    float* e2   = e1  + 16384;            // 16*512
    float* e3   = e2  + 8192;             // 16*256
    __bf16* wbf1 = (__bf16*)(e3 + 4096);  // 64*2*512 bf16 = 32768 floats
    __bf16* wbf2 = (__bf16*)(e3 + 4096 + 32768); // 1*3*512 bf16

    const int* esrc = ei;
    const int* edst = ei + E;
    const int TB = 256;
    const int egrid = (E + TB - 1) / TB;

    // ---- weight prefragmentation (bf16, WMMA B layout) ----
    pack_b_kernel<<<(64 * 2 * 512 + TB - 1) / TB, TB, 0, stream>>>(W1, wbf1, 32, 2, 64);
    pack_b_kernel<<<(1 * 3 * 512 + TB - 1) / TB, TB, 0, stream>>>(W2, wbf2, 48, 3, 1);

    // ---- conv1 ----
    gat_gemm1_wmma<<<NNODES / 32, 128, 0, stream>>>(x, wbf1, h1);
    attn_coef_kernel<<<(NNODES * 4 + TB - 1) / TB, TB, 0, stream>>>(h1, as1w, ad1w, as1, ad1, m1, 4, 8);
    edge_max_kernel<<<egrid, TB, 0, stream>>>(esrc, edst, as1, ad1, m1, E, 4);
    init_s_acc_kernel<<<(NNODES * 4 + TB - 1) / TB, TB, 0, stream>>>(h1, as1, ad1, m1, s1, hp1, 4, 8);
    edge_acc_kernel<<<egrid, TB, 0, stream>>>(esrc, edst, h1, as1, ad1, m1, s1, hp1, E, 4, 8);
    finalize_kernel<<<(NNODES * 32 + TB - 1) / TB, TB, 0, stream>>>(hp1, s1, b1, 4, 8);

    // ---- conv2 ----
    gat_gemm2_wmma<<<NNODES / 32, 192, 0, stream>>>(hp1, wbf2, h2);
    attn_coef_kernel<<<(NNODES * 3 + TB - 1) / TB, TB, 0, stream>>>(h2, as2w, ad2w, as2, ad2, m2, 3, 16);
    edge_max_kernel<<<egrid, TB, 0, stream>>>(esrc, edst, as2, ad2, m2, E, 3);
    init_s_acc_kernel<<<(NNODES * 3 + TB - 1) / TB, TB, 0, stream>>>(h2, as2, ad2, m2, s2, hp2, 3, 16);
    edge_acc_kernel<<<egrid, TB, 0, stream>>>(esrc, edst, h2, as2, ad2, m2, s2, hp2, E, 3, 16);
    finalize_kernel<<<(NNODES * 48 + TB - 1) / TB, TB, 0, stream>>>(hp2, s2, b2, 3, 16);

    // ---- pool2 -> layernorm (multiscale) -> MLP ----
    pool2_kernel<<<(NNODES + TB - 1) / TB, TB, 0, stream>>>(hp2, pool2_w, pool2_b, x2b);
    layernorm_kernel<<<16, 256, 0, stream>>>(x2b, ln0_g, ln0_b, out);                 // multiscale
    fc_kernel<<<(16 * 1024 + TB - 1) / TB, TB, 0, stream>>>(out, fc1_w, fc1_b, e1, 2048, 1024, 1);
    fc_kernel<<<(16 * 512 + TB - 1) / TB, TB, 0, stream>>>(e1, fc2_w, fc2_b, e2, 1024, 512, 1);
    fc_kernel<<<(16 * 256 + TB - 1) / TB, TB, 0, stream>>>(e2, fc3_w, fc3_b, e3, 512, 256, 1);
    fc_kernel<<<(16 * 128 + TB - 1) / TB, TB, 0, stream>>>(e3, fc4_w, fc4_b, out + 32768, 256, 128, 1); // encoded
    last_kernel<<<1, 32, 0, stream>>>(out + 32768, last_w, last_b, out + 32768 + 2048);                 // pred
}
